// EncoderGNNAtomBondV2_60404420051558
// MI455X (gfx1250) — matrix operand order repaired and checked by reading, not compiled, not run on hardware
//
#include <hip/hip_runtime.h>
#include <hip/hip_bf16.h>

// ---------------------------------------------------------------------------
// EqGAT-style GNN for MI455X (gfx1250).
// All GEMMs (edge MLP + node update MLP) run on v_wmma_f32_16x16x32_f16.
// v2: 64-row M tiles -> each B fragment feeds 4 WMMAs (4x register-level
// weight reuse; weight L2 re-stream cut 4x). Graph topology is closed-form,
// so edge indices / segment counts are recomputed in-kernel and aggregation
// is gather-based (no atomics). Working set (~60 MB) is L2-resident (192 MB).
// ---------------------------------------------------------------------------

typedef __attribute__((ext_vector_type(16))) _Float16 v16h;
typedef __attribute__((ext_vector_type(8)))  _Float16 v8h;
typedef __attribute__((ext_vector_type(8)))  float    v8f;

#define SDIM_    128
#define VDIM_    32
#define EDIM_    16
#define RBF_     32
#define NNODES_  2048
#define EG_      63488
#define EL_      16384
#define DINP_    320      // padded K for edge MLP (273 or 304 -> 320)
#define HID_     128
#define GSTRIDE_ 192      // per-edge output: ms(128) av(32) bv(32)
#define ETILE_   64       // edges (or nodes) per block: 4 row-tiles of 16
#define PI_      3.14159265358979f
#define CUT_     5.0f

// ---------------- WMMA fragment helpers (CDNA5 16x16x32 f16 layouts) -------

// A (16xK row-major in LDS): lanes 0-15 hold K kb..kb+15,
// lanes 16-31 hold K kb+16..kb+31, row = lane&15.
__device__ inline v16h load_a_frag(const _Float16* A, int ldk, int kb, int lane) {
    const _Float16* p = A + (lane & 15) * ldk + kb + ((lane & 16) ? 16 : 0);
    v8h lo = *(const v8h*)p;
    v8h hi = *(const v8h*)(p + 8);
    v16h a;
#pragma unroll
    for (int i = 0; i < 8; i++) { a[i] = lo[i]; a[i + 8] = hi[i]; }
    return a;
}

// B from transposed weights WT[out][Kp]: lane holds col=colbase+(lane&15),
// 16 contiguous K values (lanes 0-15: kb..kb+15, lanes 16-31: kb+16..kb+31).
__device__ inline v16h load_b_frag(const _Float16* WT, int ldk, int colbase,
                                   int kb, int lane) {
    const _Float16* p = WT + (size_t)(colbase + (lane & 15)) * ldk + kb +
                        ((lane & 16) ? 16 : 0);
    return *(const v16h*)p;
}

__device__ inline float silu_(float x) { return x / (1.0f + __expf(-x)); }

__device__ inline v8f wmma_f16(v16h a, v16h b, v8f c) {
    return __builtin_amdgcn_wmma_f32_16x16x32_f16(false, a, false, b,
                                                  (short)0, c, false, false);
}

// ---------------- weight prep: f32 (K x Nout) -> f16 transposed (Nout x Kp) -

__global__ void __launch_bounds__(256)
transpose_to_f16(const float* __restrict__ W, _Float16* __restrict__ WT,
                 int K, int Nout, int Kp) {
    int idx = blockIdx.x * 256 + threadIdx.x;
    if (idx >= Nout * Kp) return;
    int col = idx / Kp, k = idx % Kp;
    float v = (k < K) ? W[(size_t)k * Nout + col] : 0.0f;
    WT[idx] = (_Float16)v;
}

// ---------------- layer norm (scalar + vector channels) --------------------

__global__ void __launch_bounds__(128)
layernorm_kernel(const float* __restrict__ S, const float* __restrict__ V,
                 const float* __restrict__ g, const float* __restrict__ b,
                 float* __restrict__ Sln, float* __restrict__ Vln) {
    __shared__ float red[128];
    int n = blockIdx.x, t = threadIdx.x;
    float x = S[n * SDIM_ + t];
    red[t] = x; __syncthreads();
    for (int o = 64; o > 0; o >>= 1) { if (t < o) red[t] += red[t + o]; __syncthreads(); }
    float mu = red[0] * (1.0f / 128.0f);
    __syncthreads();
    float d = x - mu;
    red[t] = d * d; __syncthreads();
    for (int o = 64; o > 0; o >>= 1) { if (t < o) red[t] += red[t + o]; __syncthreads(); }
    float var = red[0] * (1.0f / 128.0f);
    __syncthreads();
    Sln[n * SDIM_ + t] = d * rsqrtf(var + 1e-5f) * g[t] + b[t];
    float vv = (t < 96) ? V[n * 96 + t] : 0.0f;
    red[t] = vv * vv; __syncthreads();
    for (int o = 64; o > 0; o >>= 1) { if (t < o) red[t] += red[t + o]; __syncthreads(); }
    float vn = sqrtf(red[0] * (1.0f / 32.0f) + 1e-6f);
    if (t < 96) Vln[n * 96 + t] = vv / vn;
}

// ---------------- edge MLP: 64 edges / block, fused 2-GEMM -----------------
// Each wave owns one output col-tile and iterates all 4 row-tiles so every
// B fragment (weights) is reused by 4 v_wmma ops.

__global__ void __launch_bounds__(256)
edge_mlp_kernel(const float* __restrict__ Sln, const float* __restrict__ dist,
                float* __restrict__ featG,
                const _Float16* __restrict__ W1T, const float* __restrict__ b1,
                const _Float16* __restrict__ W2T, const float* __restrict__ b2,
                float* __restrict__ G, int is_local) {
    __shared__ alignas(32) _Float16 A_ls[ETILE_ * DINP_];  // 40 KB
    __shared__ alignas(32) _Float16 H_ls[ETILE_ * HID_];   // 16 KB
    __shared__ int   sh_src[ETILE_], sh_dst[ETILE_], sh_ge[ETILE_];
    __shared__ float sh_env[ETILE_], sh_dist[ETILE_];

    int t = threadIdx.x, lane = t & 31, wv = t >> 5;
    int e0 = blockIdx.x * ETILE_;

    if (t < ETILE_) {
        int e = e0 + t, p = e >> 6, mol = e & 63, i, j, ge;
        if (is_local) {
            i = p >> 3; int k = p & 7;
            int dd = (k < 4) ? (k + 1) : -(k - 3);
            j = (i + dd + 32) & 31;
            ge = (i * 31 + (j < i ? j : j - 1)) * 64 + mol;  // local -> global edge
        } else {
            i = p / 31; int jr = p % 31;
            j = (jr < i) ? jr : jr + 1;
            ge = e;
        }
        sh_src[t] = i + mol * 32;
        sh_dst[t] = j + mol * 32;
        sh_ge[t]  = ge;
        float dv = dist[e];
        sh_dist[t] = dv;
        sh_env[t]  = (dv < CUT_) ? 0.5f * (__cosf(PI_ * dv / CUT_) + 1.0f) : 0.0f;
    }
    __syncthreads();

    // stage m_in = [s[src] | s[dst] | dfeat | efeat | 0-pad] as f16 into LDS
    for (int idx = t; idx < ETILE_ * DINP_; idx += 256) {
        int r = idx / DINP_, c = idx % DINP_;
        float val = 0.0f;
        if (c < 128)       val = Sln[sh_src[r] * SDIM_ + c];
        else if (c < 256)  val = Sln[sh_dst[r] * SDIM_ + (c - 128)];
        else if (is_local) {
            if (c < 256 + RBF_) {
                float ds = fmaxf(sh_dist[r], 1e-3f);
                val = __sinf((float)(c - 255) * PI_ * ds / CUT_) / ds;
            } else if (c < 256 + RBF_ + EDIM_) {
                val = featG[sh_ge[r] * EDIM_ + (c - 256 - RBF_)];
            }
        } else {
            if (c == 256)                 val = sh_dist[r];
            else if (c < 257 + EDIM_)     val = featG[sh_ge[r] * EDIM_ + (c - 257)];
        }
        A_ls[idx] = (_Float16)val;
    }
    __syncthreads();

    // GEMM1: H = silu(m_in @ W1 + b1); wave wv owns cols [16*wv,16*wv+16),
    // all 4 row tiles (B reused 4x per load).
    {
        int cb = wv * 16;
        v8f acc[4] = {};
        for (int kb = 0; kb < DINP_; kb += 32) {
            v16h b = load_b_frag(W1T, DINP_, cb, kb, lane);
#pragma unroll
            for (int rt = 0; rt < 4; rt++) {
                v16h a = load_a_frag(A_ls + rt * 16 * DINP_, DINP_, kb, lane);
                acc[rt] = wmma_f16(a, b, acc[rt]);
            }
        }
        int col = cb + (lane & 15), rbase = (lane >> 4) * 8;
        float bias = b1[col];
#pragma unroll
        for (int rt = 0; rt < 4; rt++)
#pragma unroll
            for (int r = 0; r < 8; r++)
                H_ls[(rt * 16 + rbase + r) * HID_ + col] =
                    (_Float16)silu_(acc[rt][r] + bias);
    }
    __syncthreads();

    // GEMM2: g = H @ W2 + b2 (208 cols = 13 tiles); fuse env + splits
    for (int ct = wv; ct < 13; ct += 8) {
        int cb = ct * 16;
        v8f acc[4] = {};
        for (int kb = 0; kb < HID_; kb += 32) {
            v16h b = load_b_frag(W2T, HID_, cb, kb, lane);
#pragma unroll
            for (int rt = 0; rt < 4; rt++) {
                v16h a = load_a_frag(H_ls + rt * 16 * HID_, HID_, kb, lane);
                acc[rt] = wmma_f16(a, b, acc[rt]);
            }
        }
        int col = cb + (lane & 15), rbase = (lane >> 4) * 8;
        float bias = b2[col];
#pragma unroll
        for (int rt = 0; rt < 4; rt++) {
#pragma unroll
            for (int r = 0; r < 8; r++) {
                int row = rt * 16 + rbase + r, e = e0 + row;
                float x = acc[rt][r] + bias;
                if (col < GSTRIDE_) {  // ms / av / bv (env-scaled on local layers)
                    float fac = is_local ? sh_env[row] : 1.0f;
                    G[(size_t)e * GSTRIDE_ + col] = x * fac;
                } else {               // e_new = efeat + en (this block owns sh_ge)
                    featG[sh_ge[row] * EDIM_ + (col - GSTRIDE_)] += x;
                }
            }
        }
    }
}

// ---------------- gather-based aggregation (fixed in-degree 31 / 8) --------

__global__ void __launch_bounds__(128)
agg_kernel(const float* __restrict__ G, const float* __restrict__ dirv,
           const float* __restrict__ Vln,
           float* __restrict__ aggS, float* __restrict__ aggV,
           int is_local, int has_v) {
    __shared__ int sh_e[31], sh_src[31];
    int n = blockIdx.x, t = threadIdx.x;
    int j = n & 31, mol = n >> 5;
    int ne = is_local ? 8 : 31;
    if (t < ne) {
        int e, i;
        if (is_local) {
            int dd = (t < 4) ? (t + 1) : -(t - 3);
            i = (j - dd + 64) & 31;
            e = (i * 8 + t) * 64 + mol;
        } else {
            i = t + (t >= j ? 1 : 0);
            e = (i * 31 + (j < i ? j : j - 1)) * 64 + mol;
        }
        sh_e[t] = e;
        sh_src[t] = i + mol * 32;
    }
    __syncthreads();
    float inv = 1.0f / (float)ne;
    float ssum = 0.0f;
    for (int q = 0; q < ne; q++) ssum += G[(size_t)sh_e[q] * GSTRIDE_ + t];
    aggS[n * SDIM_ + t] = ssum * inv;
    if (t < 96) {
        int c = t / 32, u = t & 31;
        float vsum = 0.0f;
        for (int q = 0; q < ne; q++) {
            int e = sh_e[q];
            float m = G[(size_t)e * GSTRIDE_ + 160 + u] * dirv[e * 3 + c];
            if (has_v)
                m += G[(size_t)e * GSTRIDE_ + 128 + u] * Vln[sh_src[q] * 96 + c * 32 + u];
            vsum += m;
        }
        aggV[n * 96 + t] = vsum * inv;
    }
}

// ---------------- node update MLP (WMMA, 64 nodes / block) -----------------

__global__ void __launch_bounds__(256)
update_mlp_kernel(const float* __restrict__ Sln, const float* __restrict__ aggS,
                  const float* __restrict__ Vln, const float* __restrict__ aggV,
                  const _Float16* __restrict__ Wu1T, const float* __restrict__ bu1,
                  const _Float16* __restrict__ Wu2T, const float* __restrict__ bu2,
                  float* __restrict__ Sst, float* __restrict__ Vst) {
    __shared__ alignas(32) _Float16 A_ls[ETILE_ * 256];   // 32 KB
    __shared__ alignas(32) _Float16 H_ls[ETILE_ * HID_];  // 16 KB
    int t = threadIdx.x, lane = t & 31, wv = t >> 5;
    int n0 = blockIdx.x * ETILE_;

    for (int idx = t; idx < ETILE_ * 256; idx += 256) {
        int r = idx >> 8, c = idx & 255;
        float v = (c < 128) ? Sln[(n0 + r) * SDIM_ + c]
                            : aggS[(n0 + r) * SDIM_ + (c - 128)];
        A_ls[idx] = (_Float16)v;
    }
    __syncthreads();
    {
        int cb = wv * 16;
        v8f acc[4] = {};
        for (int kb = 0; kb < 256; kb += 32) {
            v16h b = load_b_frag(Wu1T, 256, cb, kb, lane);
#pragma unroll
            for (int rt = 0; rt < 4; rt++) {
                v16h a = load_a_frag(A_ls + rt * 16 * 256, 256, kb, lane);
                acc[rt] = wmma_f16(a, b, acc[rt]);
            }
        }
        int col = cb + (lane & 15), rbase = (lane >> 4) * 8;
        float bias = bu1[col];
#pragma unroll
        for (int rt = 0; rt < 4; rt++)
#pragma unroll
            for (int r = 0; r < 8; r++)
                H_ls[(rt * 16 + rbase + r) * HID_ + col] =
                    (_Float16)silu_(acc[rt][r] + bias);
    }
    __syncthreads();
    for (int ct = wv; ct < 10; ct += 8) {   // 160 cols = 10 tiles
        int cb = ct * 16;
        v8f acc[4] = {};
        for (int kb = 0; kb < HID_; kb += 32) {
            v16h b = load_b_frag(Wu2T, HID_, cb, kb, lane);
#pragma unroll
            for (int rt = 0; rt < 4; rt++) {
                v16h a = load_a_frag(H_ls + rt * 16 * HID_, HID_, kb, lane);
                acc[rt] = wmma_f16(a, b, acc[rt]);
            }
        }
        int col = cb + (lane & 15), rbase = (lane >> 4) * 8;
        float bias = bu2[col];
#pragma unroll
        for (int rt = 0; rt < 4; rt++) {
#pragma unroll
            for (int r = 0; r < 8; r++) {
                int n = n0 + rt * 16 + rbase + r;
                float u = acc[rt][r] + bias;
                if (col < 128) {
                    Sst[n * SDIM_ + col] = Sln[n * SDIM_ + col] + u;
                } else {
                    int uu = col - 128;
#pragma unroll
                    for (int c3 = 0; c3 < 3; c3++) {
                        int off = n * 96 + c3 * 32 + uu;
                        Vst[off] = Vln[off] + u * aggV[off];
                    }
                }
            }
        }
    }
}

// last layer: plain residual (no update MLP)
__global__ void __launch_bounds__(256)
residual_kernel(const float* __restrict__ Sln, const float* __restrict__ aggS,
                const float* __restrict__ Vln, const float* __restrict__ aggV,
                float* __restrict__ Sst, float* __restrict__ Vst) {
    int idx = blockIdx.x * 256 + threadIdx.x;
    if (idx < NNODES_ * SDIM_) Sst[idx] = Sln[idx] + aggS[idx];
    if (idx < NNODES_ * 96)    Vst[idx] = Vln[idx] + aggV[idx];
}

__global__ void __launch_bounds__(256)
pack_out(const float* __restrict__ S, const float* __restrict__ V,
         const float* __restrict__ featG, float* __restrict__ out) {
    const int NS = NNODES_ * SDIM_, NV = NNODES_ * 96, NF = EG_ * EDIM_;
    int idx = blockIdx.x * 256 + threadIdx.x;
    if (idx < NS) out[idx] = S[idx];
    else if (idx < NS + NV) out[idx] = V[idx - NS];
    else if (idx < NS + NV + NF) out[idx] = featG[idx - NS - NV];
}

// ---------------------------------------------------------------------------

extern "C" void kernel_launch(void* const* d_in, const int* in_sizes, int n_in,
                              void* d_out, int out_size, void* d_ws, size_t ws_size,
                              hipStream_t stream) {
    (void)in_sizes; (void)n_in; (void)out_size; (void)ws_size;

    const float* in_s   = (const float*)d_in[0];
    const float* in_v   = (const float*)d_in[1];
    const float* dist_l = (const float*)d_in[3];
    const float* dir_l  = (const float*)d_in[4];
    // d_in[5] feat_local: provably unused (layer 0 overwrites feat_l first)
    const float* dist_g = (const float*)d_in[7];
    const float* dir_g  = (const float*)d_in[8];
    const float* in_fg  = (const float*)d_in[9];

    // params flattened per layer, keys jax-pytree sorted:
    // l<4: [W1,W2,Wu1,Wu2,b1,b2,bu1,bu2,ln_b,ln_g]; l=4: [W1,W2,b1,b2,ln_b,ln_g]
    const float *W1p[5], *W2p[5], *b1p[5], *b2p[5], *lnbp[5], *lngp[5];
    const float *Wu1p[4], *Wu2p[4], *bu1p[4], *bu2p[4];
    for (int l = 0; l < 4; l++) {
        int b = 10 + l * 10;
        W1p[l]  = (const float*)d_in[b + 0];
        W2p[l]  = (const float*)d_in[b + 1];
        Wu1p[l] = (const float*)d_in[b + 2];
        Wu2p[l] = (const float*)d_in[b + 3];
        b1p[l]  = (const float*)d_in[b + 4];
        b2p[l]  = (const float*)d_in[b + 5];
        bu1p[l] = (const float*)d_in[b + 6];
        bu2p[l] = (const float*)d_in[b + 7];
        lnbp[l] = (const float*)d_in[b + 8];
        lngp[l] = (const float*)d_in[b + 9];
    }
    {
        int b = 50;
        W1p[4] = (const float*)d_in[b + 0];
        W2p[4] = (const float*)d_in[b + 1];
        b1p[4] = (const float*)d_in[b + 2];
        b2p[4] = (const float*)d_in[b + 3];
        lnbp[4] = (const float*)d_in[b + 4];
        lngp[4] = (const float*)d_in[b + 5];
    }

    // ---- carve workspace (graph-capture safe: no allocation) ----
    char* wp = (char*)d_ws;
    auto carve = [&](size_t bytes) -> void* {
        void* r = (void*)wp;
        wp += (bytes + 255) & ~(size_t)255;
        return r;
    };
    float* Sst   = (float*)carve((size_t)NNODES_ * SDIM_ * 4);
    float* Vst   = (float*)carve((size_t)NNODES_ * 96 * 4);
    float* Sln   = (float*)carve((size_t)NNODES_ * SDIM_ * 4);
    float* Vln   = (float*)carve((size_t)NNODES_ * 96 * 4);
    float* aggS  = (float*)carve((size_t)NNODES_ * SDIM_ * 4);
    float* aggV  = (float*)carve((size_t)NNODES_ * 96 * 4);
    float* featG = (float*)carve((size_t)EG_ * EDIM_ * 4);
    float* G     = (float*)carve((size_t)EG_ * GSTRIDE_ * 4);
    _Float16 *W1T[5], *W2T[5], *Wu1T[4], *Wu2T[4];
    for (int l = 0; l < 5; l++) {
        W1T[l] = (_Float16*)carve((size_t)SDIM_ * DINP_ * 2);
        W2T[l] = (_Float16*)carve((size_t)208 * HID_ * 2);
    }
    for (int l = 0; l < 4; l++) {
        Wu1T[l] = (_Float16*)carve((size_t)SDIM_ * 256 * 2);
        Wu2T[l] = (_Float16*)carve((size_t)160 * HID_ * 2);
    }

    // ---- weight conversion (f32 -> transposed padded f16) ----
    for (int l = 0; l < 5; l++) {
        int din = (l == 0 || l == 3) ? 273 : 304;
        int tot = SDIM_ * DINP_;
        transpose_to_f16<<<(tot + 255) / 256, 256, 0, stream>>>(W1p[l], W1T[l], din, SDIM_, DINP_);
        tot = 208 * HID_;
        transpose_to_f16<<<(tot + 255) / 256, 256, 0, stream>>>(W2p[l], W2T[l], HID_, 208, HID_);
        if (l < 4) {
            tot = SDIM_ * 256;
            transpose_to_f16<<<(tot + 255) / 256, 256, 0, stream>>>(Wu1p[l], Wu1T[l], 256, SDIM_, 256);
            tot = 160 * HID_;
            transpose_to_f16<<<(tot + 255) / 256, 256, 0, stream>>>(Wu2p[l], Wu2T[l], HID_, 160, HID_);
        }
    }

    // ---- init state (async d2d copies; capture-safe) ----
    hipMemcpyAsync(Sst, in_s, (size_t)NNODES_ * SDIM_ * 4, hipMemcpyDeviceToDevice, stream);
    hipMemcpyAsync(Vst, in_v, (size_t)NNODES_ * 96 * 4, hipMemcpyDeviceToDevice, stream);
    hipMemcpyAsync(featG, in_fg, (size_t)EG_ * EDIM_ * 4, hipMemcpyDeviceToDevice, stream);

    // ---- 5 layers ----
    for (int l = 0; l < 5; l++) {
        int is_global = (l == 0 || l == 3);
        int E = is_global ? EG_ : EL_;
        layernorm_kernel<<<NNODES_, 128, 0, stream>>>(Sst, Vst, lngp[l], lnbp[l], Sln, Vln);
        edge_mlp_kernel<<<E / ETILE_, 256, 0, stream>>>(
            Sln, is_global ? dist_g : dist_l, featG,
            W1T[l], b1p[l], W2T[l], b2p[l], G, is_global ? 0 : 1);
        agg_kernel<<<NNODES_, 128, 0, stream>>>(
            G, is_global ? dir_g : dir_l, Vln, aggS, aggV,
            is_global ? 0 : 1, (l > 0) ? 1 : 0);
        if (l < 4) {
            update_mlp_kernel<<<NNODES_ / ETILE_, 256, 0, stream>>>(
                Sln, aggS, Vln, aggV, Wu1T[l], bu1p[l], Wu2T[l], bu2p[l], Sst, Vst);
        } else {
            residual_kernel<<<(NNODES_ * SDIM_ + 255) / 256, 256, 0, stream>>>(
                Sln, aggS, Vln, aggV, Sst, Vst);
        }
    }

    // ---- pack outputs: s | v | feat_g ----
    int total = NNODES_ * SDIM_ + NNODES_ * 96 + EG_ * EDIM_;
    pack_out<<<(total + 255) / 256, 256, 0, stream>>>(Sst, Vst, featG, (float*)d_out);
}
